// InMemoryRetriever_32091995636179
// MI455X (gfx1250) — matrix operand order repaired and checked by reading, not compiled, not run on hardware
//
#include <hip/hip_runtime.h>
#include <hip/hip_bf16.h>

// Problem constants (from reference setup_inputs)
#define Q_ROWS 512        // B*R = 32*16
#define DIM    768
#define NDOCS  100000
#define SEQ    128
#define TOPK   8
#define CHUNK  2048
#define NCHUNK ((NDOCS + CHUNK - 1) / CHUNK)   // 49
#define KSTEPS (DIM / 32)                       // 24 wmma k-steps
#define WAVES  8

// Output layout (floats, concatenated in reference return order)
#define OUT_TOK   0
#define OUT_MASK  (Q_ROWS * TOPK * SEQ)                 // 524288
#define OUT_SCORE (OUT_MASK + Q_ROWS * TOPK * SEQ)      // 1048576
#define OUT_IDS   (OUT_SCORE + Q_ROWS * TOPK)           // 1052672
#define OUT_EMB   (OUT_IDS + Q_ROWS * TOPK)             // 1056768

typedef __attribute__((ext_vector_type(16))) __bf16 v16bf;
typedef __attribute__((ext_vector_type(8)))  float  v8f;

union Frag { v16bf v; uint4 u[2]; };

__device__ inline unsigned short f2bf(float f) {
  unsigned int u = __float_as_uint(f);
  u += 0x7FFFu + ((u >> 16) & 1u);   // round-to-nearest-even
  return (unsigned short)(u >> 16);
}

// ---------------------------------------------------------------------------
// Kernel 1: L2-normalize rows and convert to bf16 (one wave32 per row)
// ---------------------------------------------------------------------------
__global__ void normalize_rows_bf16(const float* __restrict__ src,
                                    unsigned short* __restrict__ dst,
                                    int rows) {
  const int wave = threadIdx.x >> 5;
  const int lane = threadIdx.x & 31;
  const int row  = blockIdx.x * (blockDim.x >> 5) + wave;
  if (row >= rows) return;
  const float* r = src + (size_t)row * DIM;
  float ss = 0.f;
#pragma unroll
  for (int i = 0; i < DIM / 32; ++i) {
    float v = r[lane + i * 32];
    ss += v * v;
  }
#pragma unroll
  for (int s = 16; s >= 1; s >>= 1) ss += __shfl_xor(ss, s, 32);
  const float inv = 1.f / fmaxf(sqrtf(ss), 1e-12f);
  unsigned short* o = dst + (size_t)row * DIM;
#pragma unroll
  for (int i = 0; i < DIM / 32; ++i)
    o[lane + i * 32] = f2bf(r[lane + i * 32] * inv);
}

// ---------------------------------------------------------------------------
// Kernel 2: fused bf16-WMMA score GEMM + per-chunk top-8
//   block = (query block qb of 16 rows, doc chunk ch of 2048 docs), 8 waves
//   each wave computes 16x32 score supertiles (dual accumulators) with
//   v_wmma_f32_16x16x32_bf16; A fragments from LDS feed both WMMAs.
// ---------------------------------------------------------------------------
__global__ __launch_bounds__(256, 2)
void score_chunk_topk(const unsigned short* __restrict__ qb16,
                      const unsigned short* __restrict__ dkb16,
                      float* __restrict__ candS,
                      int* __restrict__ candI) {
  __shared__ unsigned int AsRaw[16 * DIM / 2];      // 16x768 bf16 = 24 KB
  __shared__ float stage[WAVES][16][32];            // 16 KB score staging
  __shared__ float wCandS[WAVES][16][TOPK];         // 4 KB
  __shared__ int   wCandI[WAVES][16][TOPK];         // 4 KB

  const int tid  = threadIdx.x;
  const int wave = tid >> 5;
  const int lane = tid & 31;
  const int qb   = blockIdx.x;   // 0..31
  const int ch   = blockIdx.y;   // 0..NCHUNK-1

  // cooperatively stage the 16x768 bf16 query tile into LDS
  const unsigned int* qsrc =
      (const unsigned int*)(qb16 + (size_t)qb * 16 * DIM);
  for (int i = tid; i < 16 * DIM / 2; i += 256) AsRaw[i] = qsrc[i];
  __syncthreads();
  const unsigned short* As = (const unsigned short*)AsRaw;

  const int chunkBase = ch * CHUNK;
  int docsInChunk = NDOCS - chunkBase;
  if (docsInChunk > CHUNK) docsInChunk = CHUNK;
  const int stiles = docsInChunk >> 5;   // 32-doc supertiles; NDOCS % 32 == 0

  const int  laneM = lane & 15;
  const bool hi    = lane >= 16;
  const int  kaOff = hi ? 16 : 0;   // A-frag K sub-offset (contiguous, matches B)
  const int  kbOff = hi ? 16 : 0;   // B-frag K sub-offset

  float ts[TOPK]; int ti[TOPK];
#pragma unroll
  for (int i = 0; i < TOPK; ++i) { ts[i] = -3.4e38f; ti[i] = 0; }

  const unsigned short* Arow = As + laneM * DIM;

  for (int t = wave; t < stiles; t += WAVES) {
    const int docBase = chunkBase + t * 32;
    const unsigned short* B0 = dkb16 + (size_t)(docBase + laneM) * DIM;
    const unsigned short* B1 = B0 + (size_t)16 * DIM;

    v8f acc0 = {0.f, 0.f, 0.f, 0.f, 0.f, 0.f, 0.f, 0.f};
    v8f acc1 = {0.f, 0.f, 0.f, 0.f, 0.f, 0.f, 0.f, 0.f};
#pragma unroll
    for (int s = 0; s < KSTEPS; ++s) {
      const int k0 = s * 32;
      Frag a, b0, b1;
      a.u[0]  = *(const uint4*)(Arow + k0 + kaOff);        // K=0..7 / 16..23
      a.u[1]  = *(const uint4*)(Arow + k0 + kaOff + 8);    // K=8..15 / 24..31
      b0.u[0] = *(const uint4*)(B0 + k0 + kbOff);          // K=0..7 / 16..23
      b0.u[1] = *(const uint4*)(B0 + k0 + kbOff + 8);      // K=8..15 / 24..31
      b1.u[0] = *(const uint4*)(B1 + k0 + kbOff);
      b1.u[1] = *(const uint4*)(B1 + k0 + kbOff + 8);
      acc0 = __builtin_amdgcn_wmma_f32_16x16x32_bf16(
          false, a.v, false, b0.v, (short)0, acc0, false, false);
      acc1 = __builtin_amdgcn_wmma_f32_16x16x32_bf16(
          false, a.v, false, b1.v, (short)0, acc1, false, false);
    }

    // stage C tiles: lane = column, VGPR j = row j (lo lanes) / j+8 (hi lanes)
#pragma unroll
    for (int j = 0; j < 8; ++j) {
      const int row = hi ? j + 8 : j;
      stage[wave][row][laneM]      = acc0[j];
      stage[wave][row][laneM + 16] = acc1[j];
    }
    __builtin_amdgcn_fence(__ATOMIC_ACQ_REL, "workgroup");
    __builtin_amdgcn_wave_barrier();

    if (!hi) {   // lane l owns query-row l: insert 32 candidates into top-8
#pragma unroll
      for (int col = 0; col < 32; ++col) {
        const float sc = stage[wave][laneM][col];
        if (sc > ts[TOPK - 1]) {
          ts[TOPK - 1] = sc; ti[TOPK - 1] = docBase + col;
#pragma unroll
          for (int x = TOPK - 1; x > 0; --x) {
            if (ts[x] > ts[x - 1]) {
              float tf = ts[x]; ts[x] = ts[x - 1]; ts[x - 1] = tf;
              int   tn = ti[x]; ti[x] = ti[x - 1]; ti[x - 1] = tn;
            }
          }
        }
      }
    }
    __builtin_amdgcn_fence(__ATOMIC_ACQ_REL, "workgroup");
    __builtin_amdgcn_wave_barrier();
  }

  if (!hi) {
#pragma unroll
    for (int i = 0; i < TOPK; ++i) {
      wCandS[wave][laneM][i] = ts[i];
      wCandI[wave][laneM][i] = ti[i];
    }
  }
  __syncthreads();

  // merge the 8 per-wave top-8 lists per row -> chunk top-8
  if (tid < 16) {
    float fs[TOPK]; int fi[TOPK];
#pragma unroll
    for (int i = 0; i < TOPK; ++i) { fs[i] = -3.4e38f; fi[i] = 0; }
    for (int w = 0; w < WAVES; ++w) {
#pragma unroll
      for (int i = 0; i < TOPK; ++i) {
        const float sc = wCandS[w][tid][i];
        const int   id = wCandI[w][tid][i];
        if (sc > fs[TOPK - 1]) {
          fs[TOPK - 1] = sc; fi[TOPK - 1] = id;
#pragma unroll
          for (int x = TOPK - 1; x > 0; --x) {
            if (fs[x] > fs[x - 1]) {
              float tf = fs[x]; fs[x] = fs[x - 1]; fs[x - 1] = tf;
              int   tn = fi[x]; fi[x] = fi[x - 1]; fi[x - 1] = tn;
            }
          }
        }
      }
    }
    const size_t base = (((size_t)qb * 16 + tid) * NCHUNK + ch) * TOPK;
#pragma unroll
    for (int i = 0; i < TOPK; ++i) { candS[base + i] = fs[i]; candI[base + i] = fi[i]; }
  }
}

// ---------------------------------------------------------------------------
// Kernel 3: per-row merge of NCHUNK*8 candidates -> global top-8
// ---------------------------------------------------------------------------
__global__ void final_topk(const float* __restrict__ candS,
                           const int* __restrict__ candI,
                           float* __restrict__ outScores,
                           int* __restrict__ finalIdx) {
  __shared__ float ls[32][TOPK];
  __shared__ int   li[32][TOPK];
  const int q    = blockIdx.x;
  const int lane = threadIdx.x;        // 32 threads
  const int total = NCHUNK * TOPK;     // 392
  float fs[TOPK]; int fi[TOPK];
#pragma unroll
  for (int i = 0; i < TOPK; ++i) { fs[i] = -3.4e38f; fi[i] = 0; }
  for (int i = lane; i < total; i += 32) {
    const float sc = candS[(size_t)q * total + i];
    const int   id = candI[(size_t)q * total + i];
    if (sc > fs[TOPK - 1]) {
      fs[TOPK - 1] = sc; fi[TOPK - 1] = id;
#pragma unroll
      for (int x = TOPK - 1; x > 0; --x) {
        if (fs[x] > fs[x - 1]) {
          float tf = fs[x]; fs[x] = fs[x - 1]; fs[x - 1] = tf;
          int   tn = fi[x]; fi[x] = fi[x - 1]; fi[x - 1] = tn;
        }
      }
    }
  }
#pragma unroll
  for (int i = 0; i < TOPK; ++i) { ls[lane][i] = fs[i]; li[lane][i] = fi[i]; }
  __syncthreads();
  if (lane == 0) {
    float gs[TOPK]; int gi[TOPK];
#pragma unroll
    for (int i = 0; i < TOPK; ++i) { gs[i] = -3.4e38f; gi[i] = 0; }
    for (int w = 0; w < 32; ++w) {
#pragma unroll
      for (int i = 0; i < TOPK; ++i) {
        const float sc = ls[w][i];
        const int   id = li[w][i];
        if (sc > gs[TOPK - 1]) {
          gs[TOPK - 1] = sc; gi[TOPK - 1] = id;
#pragma unroll
          for (int x = TOPK - 1; x > 0; --x) {
            if (gs[x] > gs[x - 1]) {
              float tf = gs[x]; gs[x] = gs[x - 1]; gs[x - 1] = tf;
              int   tn = gi[x]; gi[x] = gi[x - 1]; gi[x - 1] = tn;
            }
          }
        }
      }
    }
#pragma unroll
    for (int i = 0; i < TOPK; ++i) {
      outScores[(size_t)q * TOPK + i] = gs[i];
      finalIdx[(size_t)q * TOPK + i]  = gi[i];
    }
  }
}

// ---------------------------------------------------------------------------
// Kernel 4: gather tokens / mask / ids / (un-normalized) key embeddings
// ---------------------------------------------------------------------------
__global__ void gather_out(const int* __restrict__ finalIdx,
                           const int* __restrict__ docTokens,
                           const unsigned char* __restrict__ docMask,
                           const int* __restrict__ docIds,
                           const float* __restrict__ docEmb,
                           float* __restrict__ out) {
  const int p   = blockIdx.x;      // 0..Q_ROWS*TOPK-1
  const int tid = threadIdx.x;     // 256
  const int idx = finalIdx[p];
  for (int s = tid; s < SEQ; s += 256) {
    out[OUT_TOK  + (size_t)p * SEQ + s] = (float)docTokens[(size_t)idx * SEQ + s];
    out[OUT_MASK + (size_t)p * SEQ + s] = docMask[(size_t)idx * SEQ + s] ? 1.0f : 0.0f;
  }
  if (tid == 0) out[OUT_IDS + p] = (float)docIds[idx];
  for (int d = tid; d < DIM; d += 256)
    out[OUT_EMB + (size_t)p * DIM + d] = docEmb[(size_t)idx * DIM + d];
}

// ---------------------------------------------------------------------------
extern "C" void kernel_launch(void* const* d_in, const int* in_sizes, int n_in,
                              void* d_out, int out_size, void* d_ws, size_t ws_size,
                              hipStream_t stream) {
  const float* qEmb            = (const float*)d_in[0];
  const float* dkEmb           = (const float*)d_in[1];
  const int* docTok            = (const int*)d_in[2];
  const unsigned char* docMask = (const unsigned char*)d_in[3];
  const int* docIds            = (const int*)d_in[4];
  float* out                   = (float*)d_out;

  // workspace layout (~156 MB): bf16 normalized Q, bf16 normalized docs,
  // per-chunk candidates, final indices
  unsigned short* qb16  = (unsigned short*)d_ws;
  unsigned short* dkb16 = qb16 + (size_t)Q_ROWS * DIM;
  float* candS          = (float*)(dkb16 + (size_t)NDOCS * DIM);
  int*   candI          = (int*)(candS + (size_t)Q_ROWS * NCHUNK * TOPK);
  int*   finalIdx       = candI + (size_t)Q_ROWS * NCHUNK * TOPK;

  normalize_rows_bf16<<<Q_ROWS / 8, 256, 0, stream>>>(qEmb, qb16, Q_ROWS);
  normalize_rows_bf16<<<NDOCS / 8, 256, 0, stream>>>(dkEmb, dkb16, NDOCS);
  score_chunk_topk<<<dim3(32, NCHUNK), 256, 0, stream>>>(qb16, dkb16, candS, candI);
  final_topk<<<Q_ROWS, 32, 0, stream>>>(candS, candI, out + OUT_SCORE, finalIdx);
  gather_out<<<Q_ROWS * TOPK, 256, 0, stream>>>(finalIdx, docTok, docMask, docIds,
                                                dkEmb, out);
}
